// LanguageModel_43379169689963
// MI455X (gfx1250) — matrix operand order repaired and checked by reading, not compiled, not run on hardware
//
#include <hip/hip_runtime.h>
#include <cstddef>

// ---------------- Problem constants ----------------
#define VOCAB 50257
#define NPAD  50688          // 99 * 512, padded N for the output GEMM
#define DIM   512
#define HID   512
#define BATCH 16
#define SEQT  128
#define FOURH 2048           // 4*HID
#define KCAT  1024           // DIM + HID

typedef __attribute__((ext_vector_type(16))) _Float16 v16h;
typedef __attribute__((ext_vector_type(8)))  float    v8f;

union FragH {
    v16h v;
    unsigned int u[8];
};

__device__ __forceinline__ float sigm(float x) { return 1.0f / (1.0f + expf(-x)); }

// ---------------- Prep kernels ----------------

// lstm_W [1024][2048] f32 -> WT [2048][1024] f16   (B operand, K-contiguous per row)
__global__ void k_prep_lstmw(const float* __restrict__ W, _Float16* __restrict__ WT) {
    int idx = blockIdx.x * 256 + threadIdx.x;          // 2048*1024 threads
    int n = idx >> 10, k = idx & 1023;
    WT[idx] = (_Float16)W[k * FOURH + n];
}

// out_W [512][V] f32 -> OT [NPAD][512] f16 (zero-padded rows beyond V)
__global__ void k_prep_outw(const float* __restrict__ W, _Float16* __restrict__ OT) {
    const size_t total = (size_t)NPAD * 512;
    for (size_t idx = (size_t)blockIdx.x * 256 + threadIdx.x; idx < total;
         idx += (size_t)gridDim.x * 256) {
        int n = (int)(idx >> 9), k = (int)(idx & 511);
        OT[idx] = (n < VOCAB) ? (_Float16)W[(size_t)k * VOCAB + n] : (_Float16)0.0f;
    }
}

// gather embeddings into time-major f16: X[t][b][d]
__global__ void k_prep_emb(const int* __restrict__ tokens, const float* __restrict__ E,
                           _Float16* __restrict__ X) {
    int idx = blockIdx.x * 256 + threadIdx.x;          // SEQT*BATCH*DIM threads
    int t = idx >> 13;                                  // /(16*512)
    int b = (idx >> 9) & 15;
    int d = idx & 511;
    int tok = tokens[b * SEQT + t];
    X[idx] = (_Float16)E[(size_t)tok * DIM + d];
}

// zero-init c (f32) and h (f16) state
__global__ void k_init(float* __restrict__ c, _Float16* __restrict__ h) {
    int idx = blockIdx.x * 256 + threadIdx.x;          // BATCH*HID threads
    c[idx] = 0.0f;
    h[idx] = (_Float16)0.0f;
}

// ---------------- LSTM step: z = [x_t, h] @ lstm_W + b ----------------
// grid: 16 blocks x 256 threads = 128 waves; wave -> one 16x16 tile of z [16,2048]
__global__ void __launch_bounds__(256)
k_lstm_z(const _Float16* __restrict__ X,   // [T,16,512]
         const _Float16* __restrict__ H,   // [16,512]
         const _Float16* __restrict__ WT,  // [2048,1024]
         const float* __restrict__ bias,   // [2048]
         float* __restrict__ z,            // [16,2048]
         int t) {
    const int lane = threadIdx.x & 31;
    const int wave = threadIdx.x >> 5;
    const int half = lane >> 4;
    const int l16  = lane & 15;
    const int n0   = (blockIdx.x * 8 + wave) * 16;

    const _Float16* xrow = X + ((size_t)t * BATCH + l16) * DIM;   // A row (batch l16)
    const _Float16* hrow = H + (size_t)l16 * HID;
    const _Float16* brow = WT + (size_t)(n0 + l16) * KCAT + (half << 4);

    v8f acc = {};
    for (int k0 = 0; k0 < KCAT; k0 += 32) {
        FragH a, b;
        const _Float16* ap = (k0 < DIM) ? (xrow + k0) : (hrow + (k0 - DIM));
#pragma unroll
        for (int v = 0; v < 8; ++v) {
            // A 16x32 f16 operand, contiguous k order chosen here: k = half*16 + 2v
            int kl = (half << 4) + (v << 1);
            a.u[v] = *(const unsigned int*)(ap + kl);
        }
#pragma unroll
        for (int v = 0; v < 8; ++v) {
            // B 32x16 f16 operand, same contiguous k order as A: k = half*16 + 2v (row = column n of W)
            b.u[v] = *(const unsigned int*)(brow + k0 + 2 * v);
        }
        acc = __builtin_amdgcn_wmma_f32_16x16x32_f16(false, a.v, false, b.v,
                                                     (short)0, acc, false, false);
    }

    const float bv = bias[n0 + l16];
#pragma unroll
    for (int r = 0; r < 8; ++r) {
        int m = r + (half << 3);                       // C/D layout: M = r + 8*half
        z[m * FOURH + n0 + l16] = acc[r] + bv;
    }
}

// ---------------- LSTM gates + state update + masked output ----------------
__global__ void k_gates(const float* __restrict__ z, float* __restrict__ c,
                        _Float16* __restrict__ h, _Float16* __restrict__ rnn,
                        const int* __restrict__ num_words, int t) {
    int idx = blockIdx.x * 256 + threadIdx.x;          // BATCH*HID threads
    int b = idx >> 9, hh = idx & 511;
    const float* zb = z + (size_t)b * FOURH;
    float gi = zb[hh], gj = zb[HID + hh], gf = zb[2 * HID + hh], go = zb[3 * HID + hh];
    float co = c[idx];
    float nc = co * sigm(gf + 1.0f) + sigm(gi) * tanhf(gj);
    float nh = tanhf(nc) * sigm(go);
    bool m = (t < num_words[b]);
    if (m) {
        c[idx] = nc;
        h[idx] = (_Float16)nh;
    }
    rnn[((size_t)(b * SEQT + t)) * HID + hh] = m ? (_Float16)nh : (_Float16)0.0f;
}

// ---------------- Output projection: logits = rnn_out @ out_W + out_b ----------------
// block = 8 waves; A panel (16 x 512 = 16 KB) staged in LDS once per block;
// each wave computes 16(M) x 64(N) with 4 accumulators sharing one A fragment.
// grid = (NPAD/512, 2048/16)
__global__ void __launch_bounds__(256)
k_out_gemm(const _Float16* __restrict__ A,   // [2048,512]
           const _Float16* __restrict__ BT,  // [NPAD,512]
           const float* __restrict__ bias,   // [V]
           float* __restrict__ out) {        // [2048,V]
    __shared__ _Float16 As[16 * 512];                  // 16 KB A panel

    const int lane = threadIdx.x & 31;
    const int wave = threadIdx.x >> 5;
    const int half = lane >> 4;
    const int l16  = lane & 15;
    const int m0   = blockIdx.y * 16;
    const int n0   = blockIdx.x * 512 + wave * 64;

    // cooperative A-panel load: block's 16 rows are contiguous (16 KB = 1024 x b128)
    {
        const uint4* Ag  = (const uint4*)(A + (size_t)m0 * HID);
        uint4*       As4 = (uint4*)As;
#pragma unroll
        for (int j = 0; j < 4; ++j)
            As4[j * 256 + threadIdx.x] = Ag[j * 256 + threadIdx.x];
    }
    __syncthreads();

    const _Float16* arow = As + (size_t)l16 * HID;
    const _Float16* bp0  = BT + (size_t)(n0 + l16) * HID + (half << 4);

    v8f acc0 = {}, acc1 = {}, acc2 = {}, acc3 = {};
    for (int k0 = 0; k0 < HID; k0 += 32) {
        FragH a, b0, b1, b2, b3;
#pragma unroll
        for (int v = 0; v < 8; ++v) {
            int kl = (half << 4) + (v << 1);
            a.u[v] = *(const unsigned int*)(arow + k0 + kl);
        }
#pragma unroll
        for (int v = 0; v < 8; ++v) {
            b0.u[v] = *(const unsigned int*)(bp0 + k0 + 2 * v);
            b1.u[v] = *(const unsigned int*)(bp0 + 16 * HID + k0 + 2 * v);
            b2.u[v] = *(const unsigned int*)(bp0 + 32 * HID + k0 + 2 * v);
            b3.u[v] = *(const unsigned int*)(bp0 + 48 * HID + k0 + 2 * v);
        }
        acc0 = __builtin_amdgcn_wmma_f32_16x16x32_f16(false, a.v, false, b0.v,
                                                      (short)0, acc0, false, false);
        acc1 = __builtin_amdgcn_wmma_f32_16x16x32_f16(false, a.v, false, b1.v,
                                                      (short)0, acc1, false, false);
        acc2 = __builtin_amdgcn_wmma_f32_16x16x32_f16(false, a.v, false, b2.v,
                                                      (short)0, acc2, false, false);
        acc3 = __builtin_amdgcn_wmma_f32_16x16x32_f16(false, a.v, false, b3.v,
                                                      (short)0, acc3, false, false);
    }

    v8f accs[4] = {acc0, acc1, acc2, acc3};
#pragma unroll
    for (int s = 0; s < 4; ++s) {
        int n = n0 + s * 16 + l16;
        if (n < VOCAB) {
            float bv = bias[n];
#pragma unroll
            for (int r = 0; r < 8; ++r) {
                size_t row = (size_t)(m0 + r + (half << 3));
                out[row * VOCAB + n] = accs[s][r] + bv;
            }
        }
    }
}

// ---------------- Host launcher ----------------
extern "C" void kernel_launch(void* const* d_in, const int* in_sizes, int n_in,
                              void* d_out, int out_size, void* d_ws, size_t ws_size,
                              hipStream_t stream) {
    const int*   tokens    = (const int*)d_in[0];
    const int*   num_words = (const int*)d_in[1];
    const float* emb_W     = (const float*)d_in[2];
    const float* lstm_W    = (const float*)d_in[3];
    const float* lstm_b    = (const float*)d_in[4];
    const float* out_W     = (const float*)d_in[5];
    const float* out_b     = (const float*)d_in[6];
    float* out = (float*)d_out;

    // workspace carve-out (all regions 256B aligned)
    char* p = (char*)d_ws;
    _Float16* WT  = (_Float16*)p; p += (size_t)FOURH * KCAT * 2;        // 4   MiB
    _Float16* OT  = (_Float16*)p; p += (size_t)NPAD * HID * 2;          // ~50 MiB
    _Float16* Xt  = (_Float16*)p; p += (size_t)SEQT * BATCH * DIM * 2;  // 2   MiB
    _Float16* Rnn = (_Float16*)p; p += (size_t)BATCH * SEQT * HID * 2;  // 2   MiB
    _Float16* Hbf = (_Float16*)p; p += (size_t)BATCH * HID * 2;
    float*    Zws = (float*)p;    p += (size_t)BATCH * FOURH * 4;
    float*    Cst = (float*)p;    p += (size_t)BATCH * HID * 4;

    // prep: weight transposes+fp16 casts, embedding gather, state init
    k_prep_lstmw<<<dim3((FOURH * KCAT) / 256), 256, 0, stream>>>(lstm_W, WT);
    k_prep_outw <<<dim3(8192), 256, 0, stream>>>(out_W, OT);
    k_prep_emb  <<<dim3((SEQT * BATCH * DIM) / 256), 256, 0, stream>>>(tokens, emb_W, Xt);
    k_init      <<<dim3((BATCH * HID) / 256), 256, 0, stream>>>(Cst, Hbf);

    // sequential LSTM scan: 128 x (WMMA z-GEMM + fused gates)
    for (int t = 0; t < SEQT; ++t) {
        k_lstm_z<<<dim3(16), 256, 0, stream>>>(Xt, Hbf, WT, lstm_b, Zws, t);
        k_gates <<<dim3((BATCH * HID) / 256), 256, 0, stream>>>(Zws, Cst, Hbf, Rnn,
                                                                num_words, t);
    }

    // output projection [2048,512] @ [512,50257] -> d_out
    k_out_gemm<<<dim3(NPAD / 512, (BATCH * SEQT) / 16), 256, 0, stream>>>(Rnn, OT,
                                                                          out_b, out);
}
